// PiecewiseQuadratic_4458176053290
// MI455X (gfx1250) — matrix operand order repaired and checked by reading, not compiled, not run on hardware
//
#include <hip/hip_runtime.h>
#include <hip/hip_bf16.h>
#include <math.h>

// Problem constants (match reference)
#define DIM 128
#define HID 64

typedef float  v2f   __attribute__((ext_vector_type(2)));
typedef float  v4f   __attribute__((ext_vector_type(4)));
typedef float  v8f   __attribute__((ext_vector_type(8)));
typedef __bf16 v16bf __attribute__((ext_vector_type(16)));

// Workspace layout (in __bf16 elements): transposed, split weights.
#define OFF_A1H 0
#define OFF_A1L (DIM * DIM)         // 16384
#define OFF_A2H (2 * DIM * DIM)     // 32768
#define OFF_A2L (3 * DIM * DIM)     // 49152
#define OFF_W1H (4 * DIM * DIM)     // 65536
#define OFF_W1L (4 * DIM * DIM + HID * DIM)  // 73728
#define WS_ELEMS (4 * DIM * DIM + 2 * HID * DIM)  // 81920 elems = 163840 B

__device__ __forceinline__ float half16_reduce(float v) {
#pragma unroll
  for (int off = 1; off < 16; off <<= 1)
    v += __shfl_xor(v, off, 32);
  return v;
}

// ---------------------------------------------------------------------------
// Prep: split A1/A2/W1 into (hi, lo) bf16 pairs, stored transposed [col][k]
// so B-fragments are contiguous 32B runs. 160KB total -> L2-resident.
// ---------------------------------------------------------------------------
__global__ __launch_bounds__(256)
void split_weights_kernel(const float* __restrict__ A1, const float* __restrict__ A2,
                          const float* __restrict__ W1, __bf16* __restrict__ ws) {
  const int idx = blockIdx.x * 256 + threadIdx.x;   // 0 .. DIM*DIM-1
  if (idx >= DIM * DIM) return;
  const int k   = idx & (DIM - 1);
  const int col = idx >> 7;
  const float a1 = A1[(size_t)k * DIM + col];
  const float a2 = A2[(size_t)k * DIM + col];
  const __bf16 h1 = (__bf16)a1, l1 = (__bf16)(a1 - (float)h1);
  const __bf16 h2 = (__bf16)a2, l2 = (__bf16)(a2 - (float)h2);
  const size_t t = (size_t)col * DIM + k;
  ws[OFF_A1H + t] = h1;
  ws[OFF_A1L + t] = l1;
  ws[OFF_A2H + t] = h2;
  ws[OFF_A2L + t] = l2;
  if (col < HID) {
    const float w = W1[(size_t)k * HID + col];
    const __bf16 hw = (__bf16)w, lw = (__bf16)(w - (float)hw);
    ws[OFF_W1H + t] = hw;
    ws[OFF_W1L + t] = lw;
  }
}

// ---------------------------------------------------------------------------
// Main: split-precision bf16 WMMA (hi*hi + hi*lo + lo*hi), fp32 accumulate.
// One wave32 computes 16 rows. Block = 256 threads = 8 waves = 128 rows.
// ---------------------------------------------------------------------------
__global__ __launch_bounds__(256)
void pq_wmma_bf16_kernel(const float* __restrict__ x, const __bf16* __restrict__ ws,
                         const float* __restrict__ b1, const float* __restrict__ c1,
                         const float* __restrict__ b2, const float* __restrict__ c2,
                         const float* __restrict__ bW1, const float* __restrict__ W2,
                         const float* __restrict__ bW2,
                         float* __restrict__ out, int Btot) {
  const int lane  = threadIdx.x & 31;
  const int wave  = threadIdx.x >> 5;
  const int r0    = (blockIdx.x * 8 + wave) * 16;   // wave-uniform
  if (r0 >= Btot) return;                            // uniform branch, EXEC full

  const int lhalf = lane >> 4;
  const int lidx  = lane & 15;

  // ---- x tile as bf16 A-fragments (16x32, contiguous K order), hi/lo split:
  //      elems 0-15 <-> K = s*32 + lhalf*16 + e
  v16bf xh[4], xl[4];
  {
    const float* xrow = x + (size_t)(r0 + lidx) * DIM;
#pragma unroll
    for (int s = 0; s < 4; ++s) {
      const v4f* p0 = (const v4f*)(xrow + s * 32 + lhalf * 16);
      const v4f* p1 = (const v4f*)(xrow + s * 32 + lhalf * 16 + 8);
      const v4f q0 = p0[0], q1 = p0[1], q2 = p1[0], q3 = p1[1];
      float t[16];
#pragma unroll
      for (int j = 0; j < 4; ++j) {
        t[j] = q0[j]; t[4 + j] = q1[j]; t[8 + j] = q2[j]; t[12 + j] = q3[j];
      }
#pragma unroll
      for (int e = 0; e < 16; ++e) {
        const __bf16 h = (__bf16)t[e];
        xh[s][e] = h;
        xl[s][e] = (__bf16)(t[e] - (float)h);
      }
    }
  }

  const __bf16* A1h = ws + OFF_A1H;
  const __bf16* A1l = ws + OFF_A1L;
  const __bf16* A2h = ws + OFF_A2H;
  const __bf16* A2l = ws + OFF_A2L;
  const __bf16* W1h = ws + OFF_W1H;
  const __bf16* W1l = ws + OFF_W1L;

  float p1[8], p2[8], s0[8], s1[8];
#pragma unroll
  for (int v = 0; v < 8; ++v) { p1[v] = 0.f; p2[v] = 0.f; s0[v] = 0.f; s1[v] = 0.f; }

  // ======== quadratic forms: 8 column-tiles of A1/A2 ========
  for (int nt = 0; nt < 8; ++nt) {
    const int col = nt * 16 + lidx;
    v8f acc1 = {}; v8f acc2 = {};
#pragma unroll
    for (int s = 0; s < 4; ++s) {
      // B fragment 32x16: lane half <-> K half, contiguous [col][kbase..kbase+15]
      const size_t fo = (size_t)col * DIM + s * 32 + lhalf * 16;
      const v16bf bh1 = *(const v16bf*)(A1h + fo);
      const v16bf bl1 = *(const v16bf*)(A1l + fo);
      const v16bf bh2 = *(const v16bf*)(A2h + fo);
      const v16bf bl2 = *(const v16bf*)(A2l + fo);
      acc1 = __builtin_amdgcn_wmma_f32_16x16x32_bf16(false, xh[s], false, bh1, (short)0, acc1, false, false);
      acc1 = __builtin_amdgcn_wmma_f32_16x16x32_bf16(false, xh[s], false, bl1, (short)0, acc1, false, false);
      acc1 = __builtin_amdgcn_wmma_f32_16x16x32_bf16(false, xl[s], false, bh1, (short)0, acc1, false, false);
      acc2 = __builtin_amdgcn_wmma_f32_16x16x32_bf16(false, xh[s], false, bh2, (short)0, acc2, false, false);
      acc2 = __builtin_amdgcn_wmma_f32_16x16x32_bf16(false, xh[s], false, bl2, (short)0, acc2, false, false);
      acc2 = __builtin_amdgcn_wmma_f32_16x16x32_bf16(false, xl[s], false, bh2, (short)0, acc2, false, false);
    }
    // rowsum((x@A + b) .* x) in C layout (lane = N, vgpr v = M, +8 for hi half)
    const float b1v = b1[col];
    const float b2v = b2[col];
#pragma unroll
    for (int v = 0; v < 8; ++v) {
      const int m = v + lhalf * 8;
      const float xe = x[(size_t)(r0 + m) * DIM + col];
      p1[v] = fmaf(xe, acc1[v] + b1v, p1[v]);
      p2[v] = fmaf(xe, acc2[v] + b2v, p2[v]);
    }
  }

  // ======== h = relu(x@W1 + bW1) fused with s = h@W2 ========
  for (int nt = 0; nt < 4; ++nt) {
    const int col = nt * 16 + lidx;
    v8f acc = {};
#pragma unroll
    for (int s = 0; s < 4; ++s) {
      const size_t fo = (size_t)col * DIM + s * 32 + lhalf * 16;
      const v16bf bh = *(const v16bf*)(W1h + fo);
      const v16bf bl = *(const v16bf*)(W1l + fo);
      acc = __builtin_amdgcn_wmma_f32_16x16x32_bf16(false, xh[s], false, bh, (short)0, acc, false, false);
      acc = __builtin_amdgcn_wmma_f32_16x16x32_bf16(false, xh[s], false, bl, (short)0, acc, false, false);
      acc = __builtin_amdgcn_wmma_f32_16x16x32_bf16(false, xl[s], false, bh, (short)0, acc, false, false);
    }
    const float bw  = bW1[col];
    const float w20 = W2[col * 2 + 0];
    const float w21 = W2[col * 2 + 1];
#pragma unroll
    for (int v = 0; v < 8; ++v) {
      float hv = acc[v] + bw;
      hv = fmaxf(hv, 0.f);
      s0[v] = fmaf(hv, w20, s0[v]);
      s1[v] = fmaf(hv, w21, s1[v]);
    }
  }

#pragma unroll
  for (int v = 0; v < 8; ++v) {
    p1[v] = half16_reduce(p1[v]);
    p2[v] = half16_reduce(p2[v]);
    s0[v] = half16_reduce(s0[v]);
    s1[v] = half16_reduce(s1[v]);
  }

  if (lidx == 0) {
    const float c1v  = c1[0];
    const float c2v  = c2[0];
    const float bw20 = bW2[0];
    const float bw21 = bW2[1];
#pragma unroll
    for (int v = 0; v < 8; ++v) {
      const int row = r0 + v + lhalf * 8;
      const float y1 = p1[v] + c1v;
      const float y2 = p2[v] + c2v;
      const float g0 = 1.0f / (1.0f + expf(-(s0[v] + bw20)));
      const float g1 = 1.0f / (1.0f + expf(-(s1[v] + bw21)));
      out[row] = g0 * y1 + g1 * y2;
    }
  }
}

// ---------------------------------------------------------------------------
// Fallback: pure fp32 WMMA 16x16x4 (used only if ws is too small).
// ---------------------------------------------------------------------------
__global__ __launch_bounds__(256)
void pq_wmma_f32_kernel(const float* __restrict__ x,
                        const float* __restrict__ A1, const float* __restrict__ b1,
                        const float* __restrict__ c1,
                        const float* __restrict__ A2, const float* __restrict__ b2,
                        const float* __restrict__ c2,
                        const float* __restrict__ W1, const float* __restrict__ bW1,
                        const float* __restrict__ W2, const float* __restrict__ bW2,
                        float* __restrict__ out, int Btot) {
  const int lane  = threadIdx.x & 31;
  const int wave  = threadIdx.x >> 5;
  const int r0    = (blockIdx.x * 8 + wave) * 16;
  if (r0 >= Btot) return;
  const int lhalf = lane >> 4;
  const int lidx  = lane & 15;

  v2f xa[32];
  {
    const float* xrow = x + (size_t)(r0 + lidx) * DIM + lhalf * 2;
#pragma unroll
    for (int k = 0; k < 32; ++k) { xa[k].x = xrow[k * 4]; xa[k].y = xrow[k * 4 + 1]; }
  }
  float p1[8], p2[8], s0[8], s1[8];
#pragma unroll
  for (int v = 0; v < 8; ++v) { p1[v] = 0.f; p2[v] = 0.f; s0[v] = 0.f; s1[v] = 0.f; }

  for (int nt = 0; nt < 8; ++nt) {
    const int col = nt * 16 + lidx;
    v8f acc1 = {}; v8f acc2 = {};
#pragma unroll
    for (int k = 0; k < 32; ++k) {
      const int kk = k * 4 + lhalf * 2;
      v2f bf1, bf2;
      bf1.x = A1[(size_t)kk * DIM + col];
      bf1.y = A1[(size_t)(kk + 1) * DIM + col];
      bf2.x = A2[(size_t)kk * DIM + col];
      bf2.y = A2[(size_t)(kk + 1) * DIM + col];
      acc1 = __builtin_amdgcn_wmma_f32_16x16x4_f32(false, xa[k], false, bf1, (short)0, acc1, false, false);
      acc2 = __builtin_amdgcn_wmma_f32_16x16x4_f32(false, xa[k], false, bf2, (short)0, acc2, false, false);
    }
    const float b1v = b1[col];
    const float b2v = b2[col];
#pragma unroll
    for (int v = 0; v < 8; ++v) {
      const int m = v + lhalf * 8;
      const float xe = x[(size_t)(r0 + m) * DIM + col];
      p1[v] = fmaf(xe, acc1[v] + b1v, p1[v]);
      p2[v] = fmaf(xe, acc2[v] + b2v, p2[v]);
    }
  }
  for (int nt = 0; nt < 4; ++nt) {
    const int col = nt * 16 + lidx;
    v8f acc = {};
#pragma unroll
    for (int k = 0; k < 32; ++k) {
      const int kk = k * 4 + lhalf * 2;
      v2f bf;
      bf.x = W1[(size_t)kk * HID + col];
      bf.y = W1[(size_t)(kk + 1) * HID + col];
      acc = __builtin_amdgcn_wmma_f32_16x16x4_f32(false, xa[k], false, bf, (short)0, acc, false, false);
    }
    const float bw = bW1[col], w20 = W2[col * 2], w21 = W2[col * 2 + 1];
#pragma unroll
    for (int v = 0; v < 8; ++v) {
      float hv = fmaxf(acc[v] + bw, 0.f);
      s0[v] = fmaf(hv, w20, s0[v]);
      s1[v] = fmaf(hv, w21, s1[v]);
    }
  }
#pragma unroll
  for (int v = 0; v < 8; ++v) {
    p1[v] = half16_reduce(p1[v]);
    p2[v] = half16_reduce(p2[v]);
    s0[v] = half16_reduce(s0[v]);
    s1[v] = half16_reduce(s1[v]);
  }
  if (lidx == 0) {
    const float c1v = c1[0], c2v = c2[0], bw20 = bW2[0], bw21 = bW2[1];
#pragma unroll
    for (int v = 0; v < 8; ++v) {
      const int row = r0 + v + lhalf * 8;
      const float g0 = 1.0f / (1.0f + expf(-(s0[v] + bw20)));
      const float g1 = 1.0f / (1.0f + expf(-(s1[v] + bw21)));
      out[row] = g0 * (p1[v] + c1v) + g1 * (p2[v] + c2v);
    }
  }
}

extern "C" void kernel_launch(void* const* d_in, const int* in_sizes, int n_in,
                              void* d_out, int out_size, void* d_ws, size_t ws_size,
                              hipStream_t stream) {
  (void)in_sizes; (void)n_in;
  const float* x   = (const float*)d_in[0];
  const float* A1  = (const float*)d_in[1];
  const float* b1  = (const float*)d_in[2];
  const float* c1  = (const float*)d_in[3];
  const float* A2  = (const float*)d_in[4];
  const float* b2  = (const float*)d_in[5];
  const float* c2  = (const float*)d_in[6];
  const float* W1  = (const float*)d_in[7];
  const float* bW1 = (const float*)d_in[8];
  const float* W2  = (const float*)d_in[9];
  const float* bW2 = (const float*)d_in[10];
  float* out = (float*)d_out;

  const int Btot = out_size;                 // B rows
  const int grid = (Btot + 127) / 128;       // 8 waves * 16 rows per block

  if (ws_size >= (size_t)WS_ELEMS * sizeof(__bf16)) {
    __bf16* ws = (__bf16*)d_ws;
    split_weights_kernel<<<(DIM * DIM + 255) / 256, 256, 0, stream>>>(A1, A2, W1, ws);
    pq_wmma_bf16_kernel<<<grid, 256, 0, stream>>>(x, ws, b1, c1, b2, c2,
                                                  bW1, W2, bW2, out, Btot);
  } else {
    pq_wmma_f32_kernel<<<grid, 256, 0, stream>>>(x, A1, b1, c1, A2, b2, c2,
                                                 W1, bW1, W2, bW2, out, Btot);
  }
}
